// CausalSelfAttention_3856880632021
// MI455X (gfx1250) — matrix operand order repaired and checked by reading, not compiled, not run on hardware
//
#include <hip/hip_runtime.h>
#include <hip/hip_bf16.h>

#define DEV __device__ __forceinline__

typedef __attribute__((ext_vector_type(16))) __bf16 v16bf;
typedef __attribute__((ext_vector_type(8)))  float  v8f;

union FragAB {
  v16bf v;
  uint4 u[2];
  unsigned short s[16];
};

DEV unsigned short f2bf(float f) {
  unsigned int u = __float_as_uint(f);
  unsigned int r = (u + 0x7fffu + ((u >> 16) & 1u)) >> 16;
  return (unsigned short)r;
}

static constexpr int Bc  = 4;
static constexpr int T   = 2048;
static constexpr int C   = 2048;
static constexpr int NH  = 16;
static constexpr int NKV = 4;
static constexpr int HD  = 128;
static constexpr int NREP = NH / NKV;
static constexpr int KVS  = NKV * HD;  // 512, K/V row stride
static constexpr float SCALE = 0.08838834764831845f;  // 1/sqrt(128)

// ---------------------------------------------------------------------------
// f32 -> bf16 (grid-stride)
// ---------------------------------------------------------------------------
__global__ void cvt_f32_bf16(const float* __restrict__ s,
                             unsigned short* __restrict__ d, int n) {
  int i = blockIdx.x * blockDim.x + threadIdx.x;
  int stride = gridDim.x * blockDim.x;
  for (; i < n; i += stride) d[i] = f2bf(s[i]);
}

// f32 [K,N] -> bf16 transposed [N,K]; K = 1<<kshift (write-contiguous)
__global__ void cvt_tr_f32_bf16(const float* __restrict__ s,
                                unsigned short* __restrict__ d, int kshift,
                                int N) {
  int K = 1 << kshift;
  int i = blockIdx.x * blockDim.x + threadIdx.x;
  int total = K * N;
  int stride = gridDim.x * blockDim.x;
  for (; i < total; i += stride) {
    int n = i >> kshift, k = i & (K - 1);
    d[i] = f2bf(s[(size_t)k * N + n]);
  }
}

// ---------------------------------------------------------------------------
// LDS-free register-blocked bf16 WMMA GEMM.
//   A  [M,K] row-major bf16, Bt [N,K] row-major bf16 (pre-transposed weight).
//   block = 256 (8 waves): 4 waves along M x 2 along N; wave tile 32x64;
//   block tile 128x128. K-loop double-buffered; loads are immediate-offset
//   from running base pointers (no per-iter 64-bit address rebuild).
// MODE 0: f32 out [M,N]
// MODE 2: bf16 out [M,N] + fused RoPE (q/k projections)
// MODE 3: bf16 out transposed per-head [b][g][hd][t] (v projection)
// ---------------------------------------------------------------------------
template <int MODE>
__global__ __launch_bounds__(256) void gemm_bf16(
    const unsigned short* __restrict__ A, const unsigned short* __restrict__ Bt,
    void* __restrict__ OutP, const float* __restrict__ freqs,
    int M, int N, int K) {
  const int tid  = threadIdx.x;
  const int w    = tid >> 5;
  const int lane = tid & 31;
  const int half = lane >> 4;
  const int ln   = lane & 15;
  const int wm   = w & 3;
  const int wn   = w >> 2;
  const int m0   = blockIdx.y * 128 + wm * 32;
  const int n0   = blockIdx.x * 128 + wn * 64;

  const unsigned short* pa[2];
  const unsigned short* pb[4];
  for (int mi = 0; mi < 2; mi++)
    pa[mi] = A + (size_t)(m0 + mi * 16 + ln) * K + half * 16;
  for (int ni = 0; ni < 4; ni++)
    pb[ni] = Bt + (size_t)(n0 + ni * 16 + ln) * K + half * 16;

  v8f acc[2][4] = {};
  FragAB a0[2], b0[4], a1[2], b1[4];

  auto loadA = [&](FragAB* a, int ofs) {
    for (int mi = 0; mi < 2; mi++) {
      a[mi].u[0] = *(const uint4*)(pa[mi] + ofs);
      a[mi].u[1] = *(const uint4*)(pa[mi] + ofs + 8);
    }
  };
  auto loadB = [&](FragAB* bb, int ofs) {
    for (int ni = 0; ni < 4; ni++) {
      bb[ni].u[0] = *(const uint4*)(pb[ni] + ofs);
      bb[ni].u[1] = *(const uint4*)(pb[ni] + ofs + 8);
    }
  };
  auto mma = [&](FragAB* a, FragAB* bb) {
    for (int mi = 0; mi < 2; mi++)
      for (int ni = 0; ni < 4; ni++)
        acc[mi][ni] = __builtin_amdgcn_wmma_f32_16x16x32_bf16(
            false, a[mi].v, false, bb[ni].v, (short)0, acc[mi][ni], false,
            false);
  };

  loadA(a0, 0);
  loadB(b0, 0);
  const int iters = K >> 6;  // K / 64
  for (int it = 0; it < iters; ++it) {
    loadA(a1, 32);
    loadB(b1, 32);
    if (it + 2 < iters) {  // uniform prefetch two tiles ahead
      __builtin_prefetch(pa[0] + 128, 0, 1);
      __builtin_prefetch(pb[lane & 3] + 128, 0, 1);
    }
    mma(a0, b0);
    if (it + 1 < iters) {
      loadA(a0, 64);
      loadB(b0, 64);
    }
    mma(a1, b1);
    for (int mi = 0; mi < 2; mi++) pa[mi] += 64;
    for (int ni = 0; ni < 4; ni++) pb[ni] += 64;
  }

  // epilogue: C fragment -> row m = e + 8*half, col n = ln
  for (int mi = 0; mi < 2; mi++) {
    for (int ni = 0; ni < 4; ni++) {
      for (int e = 0; e < 8; e++) {
        int m = m0 + mi * 16 + e + half * 8;
        int n = n0 + ni * 16 + ln;
        float val = acc[mi][ni][e];
        if (MODE == 2) {
          // RoPE: even/odd head-dim partner is one lane over
          float p = __shfl_xor(val, 1, 32);
          int t = m & (T - 1);
          int d = n & (HD - 1);
          int i = d >> 1;
          float cs = freqs[(t * (HD / 2) + i) * 2 + 0];
          float sn = freqs[(t * (HD / 2) + i) * 2 + 1];
          val = (d & 1) ? (p * sn + val * cs) : (val * cs - p * sn);
        }
        if (MODE == 0) {
          ((float*)OutP)[(size_t)m * N + n] = val;
        } else if (MODE == 3) {
          int b = m >> 11, t = m & (T - 1);
          int g = n >> 7, d = n & (HD - 1);
          ((unsigned short*)OutP)[((size_t)((b * NKV + g) * HD + d)) * T + t] =
              f2bf(val);
        } else {
          ((unsigned short*)OutP)[(size_t)m * N + n] = f2bf(val);
        }
      }
    }
  }
}

// ---------------------------------------------------------------------------
// Causal GQA flash attention, bf16 WMMA, online softmax.
// grid = (T/128, NH, B); block = 256 (8 waves); each wave owns 16 Q rows.
// K is [b*T+s][NKV*HD] row-major; V is transposed [b][g][hd][t].
// All fragment loads are immediate offsets from two running base pointers.
// Wave-uniform control flow keeps EXEC all-ones for WMMA.
// ---------------------------------------------------------------------------
__global__ __launch_bounds__(256) void attn_fwd(
    const unsigned short* __restrict__ qb, const unsigned short* __restrict__ kb,
    const unsigned short* __restrict__ vtb, unsigned short* __restrict__ yb) {
  __shared__ __align__(16) unsigned short Pl[8][16][72];

  const int tid  = threadIdx.x;
  const int w    = tid >> 5;
  const int lane = tid & 31;
  const int half = lane >> 4;
  const int ln   = lane & 15;
  const int b    = blockIdx.z;
  const int h    = blockIdx.y;
  const int g    = h / NREP;
  const int q0   = blockIdx.x * 128 + w * 16;

  // Q fragments: 16 rows x 128
  FragAB qf[4];
  {
    const unsigned short* qp =
        qb + ((size_t)(b * T + q0 + ln)) * C + (size_t)h * HD + half * 16;
    for (int kc = 0; kc < 4; kc++) {
      qf[kc].u[0] = *(const uint4*)(qp + kc * 32);
      qf[kc].u[1] = *(const uint4*)(qp + kc * 32 + 8);
    }
  }

  float m_run[8], l_run[8];
  v8f o[8] = {};
  for (int e = 0; e < 8; e++) { m_run[e] = -1e30f; l_run[e] = 0.f; }

  // running base pointers (bumped once per s-block)
  const unsigned short* kptr =
      kb + ((size_t)(b * T + ln) * NKV + g) * HD + half * 16;
  const unsigned short* vptr =
      vtb + ((size_t)(b * NKV + g) * HD + ln) * T + half * 16;

  const int smax = q0 + 15;
  for (int s0 = 0; s0 <= smax; s0 += 64) {
    if (s0 + 64 <= smax) {  // uniform prefetch of next K/V s-block
      __builtin_prefetch(kptr + (64 + (lane & 15) * 16) * KVS, 0, 1);
      __builtin_prefetch(vptr + (size_t)(lane * 4) * T + 64, 0, 1);
    }

    // ---- S = Q K^T : per n-tile load all 4 K frags, then 4 wmmas ----
    v8f sc[4] = {};
    for (int nt = 0; nt < 4; nt++) {
      FragAB kf[4];
      for (int kc = 0; kc < 4; kc++) {
        kf[kc].u[0] = *(const uint4*)(kptr + nt * 16 * KVS + kc * 32);
        kf[kc].u[1] = *(const uint4*)(kptr + nt * 16 * KVS + kc * 32 + 8);
      }
      for (int kc = 0; kc < 4; kc++)
        sc[nt] = __builtin_amdgcn_wmma_f32_16x16x32_bf16(
            false, qf[kc].v, false, kf[kc].v, (short)0, sc[nt], false, false);
    }

    // ---- scale + causal mask ----
    float pv[4][8];
    for (int nt = 0; nt < 4; nt++) {
      int scol = s0 + nt * 16 + ln;
      for (int e = 0; e < 8; e++) {
        int row = q0 + e + half * 8;
        float x = sc[nt][e] * SCALE;
        pv[nt][e] = (scol <= row) ? x : -1e30f;
      }
    }

    // ---- online softmax (row lives across a 16-lane half) ----
    float nm[8], corr[8], ps[8];
    for (int e = 0; e < 8; e++) {
      float mx = fmaxf(fmaxf(pv[0][e], pv[1][e]), fmaxf(pv[2][e], pv[3][e]));
      for (int off = 1; off < 16; off <<= 1)
        mx = fmaxf(mx, __shfl_xor(mx, off, 32));
      nm[e]   = fmaxf(m_run[e], mx);
      corr[e] = __expf(m_run[e] - nm[e]);
      m_run[e] = nm[e];
      ps[e] = 0.f;
    }
    for (int nt = 0; nt < 4; nt++) {
      for (int e = 0; e < 8; e++) {
        float p = __expf(pv[nt][e] - nm[e]);
        ps[e] += p;
        Pl[w][e + half * 8][nt * 16 + ln] = f2bf(p);
      }
    }
    for (int e = 0; e < 8; e++) {
      float s = ps[e];
      for (int off = 1; off < 16; off <<= 1) s += __shfl_xor(s, off, 32);
      l_run[e] = l_run[e] * corr[e] + s;
    }
    for (int j = 0; j < 8; j++)
      for (int e = 0; e < 8; e++) o[j][e] *= corr[e];

    // ---- O += P @ V : batch all 8 V frags per s-chunk, then 8 wmmas ----
    for (int ch = 0; ch < 2; ch++) {
      FragAB pf;
      int kc2 = ch * 32 + half * 16;
      pf.u[0] = *(const uint4*)&Pl[w][ln][kc2];
      pf.u[1] = *(const uint4*)&Pl[w][ln][kc2 + 8];
      FragAB vf[8];
      for (int j = 0; j < 8; j++) {
        vf[j].u[0] = *(const uint4*)(vptr + j * 16 * T + ch * 32);
        vf[j].u[1] = *(const uint4*)(vptr + j * 16 * T + ch * 32 + 8);
      }
      for (int j = 0; j < 8; j++)
        o[j] = __builtin_amdgcn_wmma_f32_16x16x32_bf16(
            false, pf.v, false, vf[j].v, (short)0, o[j], false, false);
    }

    kptr += 64 * KVS;
    vptr += 64;
  }

  // ---- normalize + write y (bf16, row-major [b*T+t][h*HD+d]) ----
  float inv[8];
  for (int e = 0; e < 8; e++) inv[e] = 1.0f / l_run[e];
  for (int j = 0; j < 8; j++) {
    for (int e = 0; e < 8; e++) {
      int trow = q0 + e + half * 8;
      yb[((size_t)(b * T + trow)) * C + h * HD + j * 16 + ln] =
          f2bf(o[j][e] * inv[e]);
    }
  }
}

// ---------------------------------------------------------------------------
// Host-side launcher
// ---------------------------------------------------------------------------
extern "C" void kernel_launch(void* const* d_in, const int* in_sizes, int n_in,
                              void* d_out, int out_size, void* d_ws,
                              size_t ws_size, hipStream_t stream) {
  const float* x     = (const float*)d_in[0];
  const float* freqs = (const float*)d_in[1];
  const float* wq    = (const float*)d_in[2];
  const float* wk    = (const float*)d_in[3];
  const float* wv    = (const float*)d_in[4];
  const float* wo    = (const float*)d_in[5];

  char* ws = (char*)d_ws;
  size_t off = 0;
  auto alloc = [&](size_t bytes) -> void* {
    void* p = ws + off;
    off = (off + bytes + 255) & ~(size_t)255;
    return p;
  };

  const int M  = Bc * T;   // 8192
  const int NK = KVS;      // 512
  unsigned short* xb  = (unsigned short*)alloc((size_t)M * C * 2);
  unsigned short* wqt = (unsigned short*)alloc((size_t)C * C * 2);
  unsigned short* wkt = (unsigned short*)alloc((size_t)C * NK * 2);
  unsigned short* wvt = (unsigned short*)alloc((size_t)C * NK * 2);
  unsigned short* wot = (unsigned short*)alloc((size_t)C * C * 2);
  unsigned short* qbf = (unsigned short*)alloc((size_t)M * C * 2);
  unsigned short* kbf = (unsigned short*)alloc((size_t)M * NK * 2);
  unsigned short* vtb = (unsigned short*)alloc((size_t)M * NK * 2);
  unsigned short* ybf = (unsigned short*)alloc((size_t)M * C * 2);

  cvt_f32_bf16<<<2048, 256, 0, stream>>>(x, xb, M * C);
  cvt_tr_f32_bf16<<<2048, 256, 0, stream>>>(wq, wqt, 11, C);
  cvt_tr_f32_bf16<<<2048, 256, 0, stream>>>(wk, wkt, 11, NK);
  cvt_tr_f32_bf16<<<2048, 256, 0, stream>>>(wv, wvt, 11, NK);
  cvt_tr_f32_bf16<<<2048, 256, 0, stream>>>(wo, wot, 11, C);

  gemm_bf16<2><<<dim3(C / 128, M / 128), 256, 0, stream>>>(xb, wqt, qbf, freqs,
                                                           M, C, C);
  gemm_bf16<2><<<dim3(NK / 128, M / 128), 256, 0, stream>>>(xb, wkt, kbf,
                                                            freqs, M, NK, C);
  gemm_bf16<3><<<dim3(NK / 128, M / 128), 256, 0, stream>>>(xb, wvt, vtb,
                                                            nullptr, M, NK, C);

  attn_fwd<<<dim3(T / 128, NH, Bc), 256, 0, stream>>>(qbf, kbf, vtb, ybf);

  gemm_bf16<0><<<dim3(C / 128, M / 128), 256, 0, stream>>>(ybf, wot, d_out,
                                                           nullptr, M, C, C);
}
